// FullAttention_88244398064348
// MI455X (gfx1250) — matrix operand order repaired and checked by reading, not compiled, not run on hardware
//
#include <hip/hip_runtime.h>

typedef __attribute__((ext_vector_type(16))) _Float16 v16h;
typedef __attribute__((ext_vector_type(8)))  _Float16 v8h;
typedef __attribute__((ext_vector_type(2)))  _Float16 v2h;
typedef __attribute__((ext_vector_type(8)))  float    v8f;
typedef __attribute__((ext_vector_type(4)))  float    v4f;

namespace {
constexpr int kB = 2;
constexpr int kL = 2048;
constexpr int kH = 16;
constexpr int kE = 64;
constexpr int BM = 64;          // query rows per workgroup
constexpr int BN = 64;          // kv rows per iteration
constexpr int NWAVE = 4;
constexpr int LDSS = 72;        // padded LDS row stride in halfs (144B, 16B-aligned)
constexpr int SROW = kH * kE;   // fp32 elements between consecutive seq rows
}

// Stage one KV tile: K row-major f16 (packed b128 stores), V transposed f16
// (row-pair packed b32 stores). 128 threads cooperate.
__device__ __forceinline__ void stage_tile(const float* __restrict__ Kbh,
                                           const float* __restrict__ Vbh,
                                           int sbase, int nleft,
                                           _Float16 (*KSb)[LDSS],
                                           _Float16 (*VTb)[LDSS], int tid) {
  {   // K: thread -> (row, 32-wide e chunk); 16 cvt_pk + 4 ds_store_b128
    const int srow = tid >> 1;
    const int e0   = (tid & 1) * 32;
    const float* kp = Kbh + (size_t)(sbase + srow) * SROW + e0;
    #pragma unroll
    for (int u = 0; u < 4; ++u) {
      v4f x0 = *(const v4f*)(kp + u * 8);
      v4f x1 = *(const v4f*)(kp + u * 8 + 4);
      v8h hh;
      #pragma unroll
      for (int i = 0; i < 4; ++i) { hh[i] = (_Float16)x0[i]; hh[4 + i] = (_Float16)x1[i]; }
      *(v8h*)&KSb[srow][e0 + u * 8] = hh;
    }
    if (nleft > 1) __builtin_prefetch(kp + (size_t)BN * SROW, 0, 1);
  }
  {   // V: thread -> (row pair 2p,2p+1; 16-wide d chunk); 16 cvt_pk + 16 b32 st
    const int sp = tid >> 2;
    const int d0 = (tid & 3) * 16;
    const float* vp0 = Vbh + (size_t)(sbase + 2 * sp) * SROW + d0;
    const float* vp1 = vp0 + SROW;
    #pragma unroll
    for (int u = 0; u < 4; ++u) {
      v4f a = *(const v4f*)(vp0 + u * 4);
      v4f c = *(const v4f*)(vp1 + u * 4);
      #pragma unroll
      for (int i = 0; i < 4; ++i) {
        v2h pk; pk[0] = (_Float16)a[i]; pk[1] = (_Float16)c[i];
        *(v2h*)&VTb[d0 + u * 4 + i][2 * sp] = pk;
      }
    }
    if (nleft > 1) __builtin_prefetch(vp0 + (size_t)BN * SROW, 0, 1);
  }
}

// One KV tile: S^T = K @ Q^T, online softmax (log2 domain), O^T += V^T @ P^T.
// All A-fragments are preloaded before their WMMA bursts so ds loads pipeline
// with partial s_wait_dscnt instead of a serialized load->wait->wmma chain.
template <bool MASKED>
__device__ __forceinline__ void process_tile(int sbase,
                                             const _Float16 (*KSb)[LDSS],
                                             const _Float16 (*VTb)[LDSS],
                                             _Float16 (*PTw)[LDSS],
                                             const v16h (&bq)[2], v8f (&o)[4],
                                             float& m_i, float& l_i,
                                             int qrow, int ln, int hi, int kb) {
  // ---- preload all 8 K A-fragments (16 ds_load_b128 in flight) ----
  v16h ak[4][2];
  #pragma unroll
  for (int t = 0; t < 4; ++t) {
    const _Float16* krow = &KSb[t * 16 + ln][0];
    #pragma unroll
    for (int c = 0; c < 2; ++c) {
      v8h a0 = *(const v8h*)(krow + c * 32 + kb);
      v8h a1 = *(const v8h*)(krow + c * 32 + 8 + kb);
      #pragma unroll
      for (int i = 0; i < 8; ++i) { ak[t][c][i] = a0[i]; ak[t][c][8 + i] = a1[i]; }
    }
  }

  // ---- S^T: 8 back-to-back WMMAs ----
  v8f s[4];
  #pragma unroll
  for (int t = 0; t < 4; ++t) {
    v8f acc;
    #pragma unroll
    for (int r = 0; r < 8; ++r) acc[r] = 0.0f;
    #pragma unroll
    for (int c = 0; c < 2; ++c)
      acc = __builtin_amdgcn_wmma_f32_16x16x32_f16(false, ak[t][c], false, bq[c],
                                                   (short)0, acc, false, false);
    s[t] = acc;
  }

  // ---- preload all 8 V^T A-fragments now: independent of softmax, so these
  //      ds loads fly during the whole VALU/transcendental stretch below ----
  v16h av[4][2];
  #pragma unroll
  for (int td = 0; td < 4; ++td) {
    const _Float16* vrow = &VTb[td * 16 + ln][0];
    #pragma unroll
    for (int c = 0; c < 2; ++c) {
      v8h a0 = *(const v8h*)(vrow + c * 32 + kb);
      v8h a1 = *(const v8h*)(vrow + c * 32 + 8 + kb);
      #pragma unroll
      for (int i = 0; i < 8; ++i) { av[td][c][i] = a0[i]; av[td][c][8 + i] = a1[i]; }
    }
  }

  // ---- (mask +) online softmax, log2 domain; scalar stats per lane ----
  float mx = m_i;
  #pragma unroll
  for (int t = 0; t < 4; ++t)
    #pragma unroll
    for (int r = 0; r < 8; ++r) {
      if (MASKED) {
        const int sg = sbase + t * 16 + hi * 8 + r;
        const float v = (sg <= qrow) ? s[t][r] : -1e30f;
        s[t][r] = v;
      }
      mx = fmaxf(mx, s[t][r]);
    }
  mx = fmaxf(mx, __shfl_xor(mx, 16, 32));
  const float alpha = __builtin_amdgcn_exp2f(m_i - mx);
  m_i = mx;
  float sum = 0.0f;
  #pragma unroll
  for (int t = 0; t < 4; ++t)
    #pragma unroll
    for (int r = 0; r < 8; ++r) {
      const float p = __builtin_amdgcn_exp2f(s[t][r] - mx);
      s[t][r] = p;
      sum += p;
    }
  sum += __shfl_xor(sum, 16, 32);
  l_i = l_i * alpha + sum;
  #pragma unroll
  for (int t = 0; t < 4; ++t)
    #pragma unroll
    for (int r = 0; r < 8; ++r) o[t][r] *= alpha;

  // ---- P^T round trip through per-wave LDS (C-layout -> B-layout) ----
  #pragma unroll
  for (int t = 0; t < 4; ++t) {
    v8h ph;
    #pragma unroll
    for (int r = 0; r < 8; ++r) ph[r] = (_Float16)s[t][r];
    *(v8h*)&PTw[ln][t * 16 + hi * 8] = ph;
  }
  asm volatile("s_wait_dscnt 0x0" ::: "memory");   // intra-wave store->load

  v16h bp[2];
  #pragma unroll
  for (int c = 0; c < 2; ++c) {
    v8h p0 = *(const v8h*)&PTw[ln][c * 32 + hi * 16];
    v8h p1 = *(const v8h*)&PTw[ln][c * 32 + hi * 16 + 8];
    #pragma unroll
    for (int i = 0; i < 8; ++i) { bp[c][i] = p0[i]; bp[c][8 + i] = p1[i]; }
  }

  // ---- O^T += V^T @ P^T: 8 WMMAs, all operands already in VGPRs ----
  #pragma unroll
  for (int c = 0; c < 2; ++c)
    #pragma unroll
    for (int td = 0; td < 4; ++td)
      o[td] = __builtin_amdgcn_wmma_f32_16x16x32_f16(false, av[td][c], false, bp[c],
                                                     (short)0, o[td], false, false);
}

// FlashAttention-2 forward, causal; transposed formulation (S^T / O^T) so each
// lane owns one query row -> scalar softmax stats, contiguous P^T LDS traffic.
__global__ __launch_bounds__(128) void fa_fwd(const float* __restrict__ Q,
                                              const float* __restrict__ K,
                                              const float* __restrict__ V,
                                              float* __restrict__ Out) {
  const int qblk = blockIdx.x;
  const int h    = blockIdx.y;
  const int b    = blockIdx.z;
  const int tid  = threadIdx.x;
  const int wave = tid >> 5;
  const int lane = tid & 31;
  const int ln   = lane & 15;
  const int hi   = lane >> 4;
  const int kb   = hi ? 16 : 0;

  __shared__ __align__(16) _Float16 KS[BN][LDSS];
  __shared__ __align__(16) _Float16 VT[kE][LDSS];
  __shared__ __align__(16) _Float16 PT[NWAVE][16][LDSS];

  const int qrow = qblk * BM + wave * 16 + ln;   // this lane's query row
  const size_t bhoff = (size_t)b * kL * SROW + (size_t)h * kE;
  const float* Kbh = K + bhoff;
  const float* Vbh = V + bhoff;

  // ---- Q^T in B-layout, pre-scaled by log2(e)/sqrt(E) (log2-domain softmax) ----
  v16h bq[2];
  {
    constexpr float QSCALE = 0.125f * 1.44269504088896340736f;
    const float* qp = Q + bhoff + (size_t)qrow * SROW + hi * 16;
    #pragma unroll
    for (int c = 0; c < 2; ++c) {
      #pragma unroll
      for (int u = 0; u < 4; ++u) {
        v4f q = *(const v4f*)(qp + c * 32 + u * 4);
        #pragma unroll
        for (int i = 0; i < 4; ++i) bq[c][u * 4 + i] = (_Float16)(q[i] * QSCALE);
      }
    }
  }

  v8f o[4];
  #pragma unroll
  for (int t = 0; t < 4; ++t)
    #pragma unroll
    for (int r = 0; r < 8; ++r) o[t][r] = 0.0f;
  float m_i = -1e30f, l_i = 0.0f;

  for (int j = 0; j <= qblk; ++j) {
    __syncthreads();   // protect previous iteration's KS/VT reads
    stage_tile(Kbh, Vbh, j * BN, qblk + 1 - j, KS, VT, tid);
    __syncthreads();
    if (j < qblk)
      process_tile<false>(j * BN, KS, VT, PT[wave], bq, o, m_i, l_i,
                          qrow, ln, hi, kb);
    else               // diagonal tile: the only one that needs causal masking
      process_tile<true>(j * BN, KS, VT, PT[wave], bq, o, m_i, l_i,
                         qrow, ln, hi, kb);
  }

  // ---- normalize and store: lane owns row qrow; contiguous 8-float runs ----
  const float inv = 1.0f / l_i;
  float* op = Out + bhoff + (size_t)qrow * SROW + hi * 8;
  #pragma unroll
  for (int td = 0; td < 4; ++td) {
    v4f lo, hi4;
    #pragma unroll
    for (int i = 0; i < 4; ++i) { lo[i] = o[td][i] * inv; hi4[i] = o[td][4 + i] * inv; }
    *(v4f*)(op + td * 16)     = lo;
    *(v4f*)(op + td * 16 + 4) = hi4;
  }
}

extern "C" void kernel_launch(void* const* d_in, const int* in_sizes, int n_in,
                              void* d_out, int out_size, void* d_ws, size_t ws_size,
                              hipStream_t stream) {
  (void)in_sizes; (void)n_in; (void)out_size; (void)d_ws; (void)ws_size;
  const float* Q = (const float*)d_in[0];
  const float* K = (const float*)d_in[1];
  const float* V = (const float*)d_in[2];
  float* O = (float*)d_out;
  dim3 grid(kL / BM, kH, kB);   // (q-blocks, heads, batch)
  fa_fwd<<<grid, dim3(128), 0, stream>>>(Q, K, V, O);
}
